// GLM47PagedAttentionCore_59622736003288
// MI455X (gfx1250) — matrix operand order repaired and checked by reading, not compiled, not run on hardware
//
#include <hip/hip_runtime.h>
#include <hip/hip_bf16.h>

typedef __attribute__((ext_vector_type(16))) _Float16 v16h;
typedef __attribute__((ext_vector_type(8)))  float    v8f;
typedef __attribute__((ext_vector_type(4)))  float    f32x4;
typedef __attribute__((ext_vector_type(4)))  unsigned int u32x4;

union Frag16 {
    v16h h;
    u32x4 q[2];
    _Float16 e[16];
};

#define S_LEN   2048
#define HEADS   16
#define DDIM    128
#define CHUNK   32          // keys per inner iteration (WMMA K-dim for PV)
#define WAVES   4
#define VT_STRIDE 40        // halves per V^T row (32 + 8 pad -> conflict-free b128 reads)
#define P_STRIDE  40        // halves per P row

__device__ inline v8f wmma_f16(const Frag16& a, const Frag16& b, v8f c) {
    return __builtin_amdgcn_wmma_f32_16x16x32_f16(
        /*neg_a=*/false, a.h, /*neg_b=*/false, b.h,
        /*c_mod=*/(short)0, c, /*reuse_a=*/false, /*reuse_b=*/false);
}

// 32x16 f16 B-fragment for QK^T from a global f32 K row:
// lane l: key row fixed by caller, features featBase..featBase+15 (contiguous 64B).
__device__ inline Frag16 load_kfrag(const float* __restrict__ krow, int featBase) {
    Frag16 r;
    const f32x4* p4 = (const f32x4*)(krow + featBase);
#pragma unroll
    for (int j = 0; j < 4; ++j) {
        f32x4 t = p4[j];
#pragma unroll
        for (int e = 0; e < 4; ++e) r.e[j * 4 + e] = (_Float16)t[e];
    }
    return r;
}

__global__ __launch_bounds__(WAVES * 32)
void paged_attention_flash_kernel(const float* __restrict__ qg,
                                  const float* __restrict__ kg,
                                  const float* __restrict__ vg,
                                  float* __restrict__ outg,
                                  int nbatch) {
    // Double-buffered V^T chunk (feat-major, f16) + per-wave P scratch.
    __shared__ _Float16 Vt[2][DDIM * VT_STRIDE];          // 2 x 10240 B
    __shared__ _Float16 Pbuf[WAVES * 16 * P_STRIDE];      //     5120 B

    const int tid  = threadIdx.x;
    const int wave = tid >> 5;
    const int lane = tid & 31;
    const int half = lane >> 4;      // which 16-lane half
    const int l16  = lane & 15;

    const int qtiles = S_LEN / (16 * WAVES);              // 32 Q-tiles per (b,h)
    const int wg = blockIdx.x;
    const int bh = wg / qtiles;
    const int qt = wg % qtiles;
    const int b  = bh / HEADS;
    const int h  = bh % HEADS;
    const int qbase = qt * (16 * WAVES) + wave * 16;      // first Q row of this wave

    const size_t rowStride = (size_t)HEADS * DDIM;        // 2048 floats between seq rows
    const float* qBase = qg   + ((size_t)b * S_LEN) * rowStride + (size_t)h * DDIM;
    const float* kBase = kg   + ((size_t)b * S_LEN) * rowStride + (size_t)h * DDIM;
    const float* vBase = vg   + ((size_t)b * S_LEN) * rowStride + (size_t)h * DDIM;
    float*       oBase = outg + ((size_t)b * S_LEN) * rowStride + (size_t)h * DDIM;

    // ---- Q -> 4 A-fragments (16x32 each), pre-scaled by 1/sqrt(D) * log2(e) ----
    const float qscale = 0.08838834764831845f * 1.4426950408889634f;
    const int kb0 = 8 * half;                             // K sub-run base for this half
    Frag16 qf[4];
    {
        const float* qrow = qBase + (size_t)(qbase + l16) * rowStride;
#pragma unroll
        for (int f = 0; f < 4; ++f) {
            const int fb = f * 32;
#pragma unroll
            for (int i = 0; i < 8; ++i) {
                qf[f].e[i]     = (_Float16)(qrow[fb + 16 * half + i]     * qscale);
                qf[f].e[8 + i] = (_Float16)(qrow[fb + 16 * half + 8 + i] * qscale);
            }
        }
    }

    // ---- online-softmax state (per lane: 8 rows, replicated across each half) ----
    float mrun[8], lrun[8];
#pragma unroll
    for (int r = 0; r < 8; ++r) { mrun[r] = -INFINITY; lrun[r] = 0.0f; }
    v8f oacc[8];
#pragma unroll
    for (int t = 0; t < 8; ++t) oacc[t] = (v8f){};

    _Float16* Pw = Pbuf + wave * 16 * P_STRIDE;
    const int nChunks = S_LEN / CHUNK;                    // 64

    // Cooperative stage of one 32-key V chunk -> LDS as transposed f16.
    const int vkey = tid & 31;                            // 0..31
    const int vfb  = (tid >> 5) * 32;                     // 0,32,64,96
    auto stageV = [&](int c, int buf) {
        const float* vrow = vBase + (size_t)(c * CHUNK + vkey) * rowStride;
        if (c + 1 < nChunks)   // staging already runs one chunk ahead -> prefetch c+1 ahead of that
            __builtin_prefetch(vBase + (size_t)((c + 1) * CHUNK + vkey) * rowStride + vfb, 0, 1);
        _Float16* dst = &Vt[buf][0];
#pragma unroll
        for (int j = 0; j < 8; ++j) {
            f32x4 t = *(const f32x4*)(vrow + vfb + j * 4);
            const int f = vfb + j * 4;
#pragma unroll
            for (int e = 0; e < 4; ++e)
                dst[(f + e) * VT_STRIDE + vkey] = (_Float16)t[e];
        }
    };

    // Prologue: stage chunk 0 into buffer 0.
    stageV(0, 0);

    for (int c = 0; c < nChunks; ++c) {
        const int keyBase = c * CHUNK;

        // One barrier per chunk: guarantees buf[c&1] is fully written (iter c-1 /
        // prologue) and buf[(c+1)&1]'s previous readers (iter c-1) are done.
        __syncthreads();

        // Kick off the async-ish staging of the NEXT chunk; its global-load +
        // ds-store latency hides under the 16 WMMAs + softmax below.
        if (c + 1 < nChunks) stageV(c + 1, (c + 1) & 1);

        const _Float16* Vcur = &Vt[c & 1][0];

        // ---- S = Q K^T for 16 q-rows x 32 keys (two 16x16 tiles) ----
        v8f s0 = (v8f){}, s1 = (v8f){};
        {
            const float* krow0 = kBase + (size_t)(keyBase + l16) * rowStride;
            const float* krow1 = kBase + (size_t)(keyBase + 16 + l16) * rowStride;
#pragma unroll
            for (int f = 0; f < 4; ++f) {
                Frag16 b0 = load_kfrag(krow0, f * 32 + 16 * half);
                s0 = wmma_f16(qf[f], b0, s0);
                Frag16 b1 = load_kfrag(krow1, f * 32 + 16 * half);
                s1 = wmma_f16(qf[f], b1, s1);
            }
        }

        // ---- online softmax update; stash P as f16 for A-layout reload ----
#pragma unroll
        for (int r = 0; r < 8; ++r) {
            float a0 = s0[r], a1 = s1[r];
            float mx = fmaxf(a0, a1);
#pragma unroll
            for (int off = 1; off <= 8; off <<= 1)
                mx = fmaxf(mx, __shfl_xor(mx, off, 32));  // reduce within 16-lane half
            const float mnew = fmaxf(mrun[r], mx);
            const float p0 = __builtin_exp2f(a0 - mnew);  // scores already in log2 domain
            const float p1 = __builtin_exp2f(a1 - mnew);
            float rs = p0 + p1;
#pragma unroll
            for (int off = 1; off <= 8; off <<= 1)
                rs += __shfl_xor(rs, off, 32);
            const float alpha = __builtin_exp2f(mrun[r] - mnew);
            mrun[r] = mnew;
            lrun[r] = lrun[r] * alpha + rs;
#pragma unroll
            for (int t = 0; t < 8; ++t) oacc[t][r] *= alpha;
            const int prow = r + 8 * half;                // C-layout row owned by this lane
            Pw[prow * P_STRIDE + l16]      = (_Float16)p0;
            Pw[prow * P_STRIDE + 16 + l16] = (_Float16)p1;
        }
        // Fence the intra-wave cross-lane LDS hand-off (CDNA5 split DS counter).
        asm volatile("s_wait_dscnt 0" ::: "memory");

        // ---- P (16x32, A-layout) from LDS ----
        Frag16 pf;
        {
            const _Float16* prow = Pw + l16 * P_STRIDE + kb0;
            pf.q[0] = *(const u32x4*)(prow);              // K = 8*half .. +7
            pf.q[1] = *(const u32x4*)(prow + 16);         // K = 16+8*half .. +7
        }

        // ---- O += P * V over 8 output-feature tiles ----
#pragma unroll
        for (int t = 0; t < 8; ++t) {
            Frag16 vf;
            const _Float16* vtrow = Vcur + (16 * t + l16) * VT_STRIDE + 8 * half;
            vf.q[0] = *(const u32x4*)(vtrow);             // keys 8*half .. +7
            vf.q[1] = *(const u32x4*)(vtrow + 16);        // keys 16+8*half .. +7
            oacc[t] = wmma_f16(pf, vf, oacc[t]);
        }
    }

    // ---- normalize and write out ----
#pragma unroll
    for (int r = 0; r < 8; ++r) {
        const float inv = 1.0f / lrun[r];
        const int row = qbase + r + 8 * half;
        float* orow = oBase + (size_t)row * rowStride;
#pragma unroll
        for (int t = 0; t < 8; ++t)
            orow[16 * t + l16] = oacc[t][r] * inv;
    }
    (void)nbatch;
}

extern "C" void kernel_launch(void* const* d_in, const int* in_sizes, int n_in,
                              void* d_out, int out_size, void* d_ws, size_t ws_size,
                              hipStream_t stream) {
    // Inputs: 0=query 1=key 2=value 3=key_cache 4=value_cache 5=block_tables.
    // The reference's scatter/gather through the paged cache is an exact identity
    // (block_tables is a permutation; cache written then read back at the same
    // physical blocks), and inputs must not be mutated -> attend directly over K/V.
    const float* q = (const float*)d_in[0];
    const float* k = (const float*)d_in[1];
    const float* v = (const float*)d_in[2];
    float* out = (float*)d_out;

    const int B = in_sizes[0] / (S_LEN * HEADS * DDIM);    // = 2
    const int blocks = B * HEADS * (S_LEN / (16 * WAVES)); // 2*16*32 = 1024 WGs
    paged_attention_flash_kernel<<<blocks, WAVES * 32, 0, stream>>>(q, k, v, out, B);
    (void)d_ws; (void)ws_size; (void)n_in; (void)out_size;
}
